// RNNModel_76166950028140
// MI455X (gfx1250) — matrix operand order repaired and checked by reading, not compiled, not run on hardware
//
#include <hip/hip_runtime.h>
#include <hip/hip_bf16.h>

#define T_  1024
#define B_  64
#define D_  256
#define H_  512
#define OUT_ 256
#define NWG 32          // persistent workgroups; each owns 16 h-columns

typedef __attribute__((ext_vector_type(16))) __bf16 v16bf;
typedef __attribute__((ext_vector_type(8)))  __bf16 v8bf;
typedef __attribute__((ext_vector_type(8)))  float  v8f;

// ---------------------------------------------------------------------------
// fragment helpers (16x16x32 bf16 WMMA, wave32 layouts per CDNA5 ISA 7.12.2)
// ---------------------------------------------------------------------------
__device__ __forceinline__ v16bf frag_combine(v8bf lo, v8bf hi) {
  return __builtin_shufflevector(lo, hi, 0,1,2,3,4,5,6,7,8,9,10,11,12,13,14,15);
}
__device__ __forceinline__ v16bf load_a_frag(const __bf16* row, int k, int koff) {
  v8bf lo = *(const v8bf*)(row + k + koff);
  v8bf hi = *(const v8bf*)(row + k + koff + 16);
  return frag_combine(lo, hi);
}
__device__ __forceinline__ v16bf load_b_frag(const __bf16* col, int k, int kboff) {
  v8bf lo = *(const v8bf*)(col + k + (kboff >> 1));
  v8bf hi = *(const v8bf*)(col + k + (kboff >> 1) + 16);
  return frag_combine(lo, hi);
}

__device__ __forceinline__ float sigmoidf_(float x) {
  return 1.0f / (1.0f + __expf(-x));
}

// ---------------------------------------------------------------------------
// CDNA5 async global->LDS copy (ASYNCcnt-tracked; ISA 10.x / 15.18.3).
// lds_addr = low 32 bits of a generic shared pointer (aperture rule: LDS
// flat address keeps the LDS byte offset in addr[31:0]).
// ---------------------------------------------------------------------------
__device__ __forceinline__ void async_ld_b128(unsigned lds_addr, const __bf16* g) {
  asm volatile("global_load_async_to_lds_b128 %0, %1, off"
               :: "v"(lds_addr), "v"(g) : "memory");
}
__device__ __forceinline__ void wait_async0() {
  asm volatile("s_wait_asynccnt 0x0" ::: "memory");
}

// ---------------------------------------------------------------------------
// grid-wide sense barrier (persistent-kernel step sync)
// ---------------------------------------------------------------------------
__device__ __forceinline__ void grid_barrier(unsigned* cnt, unsigned* gen, unsigned nwg) {
  __threadfence();
  __syncthreads();
  if (threadIdx.x == 0) {
    unsigned g = *(volatile unsigned*)gen;
    unsigned old = atomicAdd(cnt, 1u);
    if (old == nwg - 1u) {
      *(volatile unsigned*)cnt = 0u;       // self-restoring for graph replay
      __threadfence();
      atomicAdd(gen, 1u);
    } else {
      while (*(volatile unsigned*)gen == g) { __builtin_amdgcn_s_sleep(2); }
    }
  }
  __syncthreads();
  __threadfence();
}

// ---------------------------------------------------------------------------
// prep kernels
// ---------------------------------------------------------------------------
__global__ void convert_x_kernel(const float* __restrict__ in, __bf16* __restrict__ out) {
  long long idx = (long long)blockIdx.x * blockDim.x + threadIdx.x;
  if (idx >= (long long)B_ * T_ * D_) return;
  int d = (int)(idx % D_);
  long long bt = idx / D_;
  int t = (int)(bt % T_);
  int b = (int)(bt / T_);
  out[((long long)t * B_ + b) * D_ + d] = (__bf16)in[idx];
}

// W [4H, K] f32 -> dst[NWG][64][K] bf16 (gate-interleaved slices)
__global__ void pack_gate_slices_kernel(const float* __restrict__ W,
                                        __bf16* __restrict__ dst, int K) {
  long long idx = (long long)blockIdx.x * blockDim.x + threadIdx.x;
  if (idx >= (long long)4 * H_ * K) return;
  int k = (int)(idx % K);
  int r = (int)(idx / K);        // r = s*64 + j
  int s = r >> 6, j = r & 63;
  int grow = (j >> 4) * H_ + s * 16 + (j & 15);
  dst[idx] = (__bf16)W[(long long)grow * K + k];
}

__global__ void pack_bias_kernel(const float* __restrict__ bih,
                                 const float* __restrict__ bhh,
                                 float* __restrict__ dst) {
  int r = blockIdx.x * blockDim.x + threadIdx.x;
  if (r >= 4 * H_) return;
  int s = r >> 6, j = r & 63;
  int grow = (j >> 4) * H_ + s * 16 + (j & 15);
  dst[r] = bih[grow] + bhh[grow];
}

__global__ void convert_fcw_kernel(const float* __restrict__ in, __bf16* __restrict__ out) {
  int idx = blockIdx.x * blockDim.x + threadIdx.x;
  if (idx >= OUT_ * H_) return;
  out[idx] = (__bf16)in[idx];
}

__global__ void zero_init_kernel(__bf16* __restrict__ h1_0,
                                 __bf16* __restrict__ h2pp,
                                 unsigned* __restrict__ bar) {
  int idx = blockIdx.x * blockDim.x + threadIdx.x;
  if (idx < B_ * H_)        h1_0[idx] = (__bf16)0.0f;
  if (idx < 2 * B_ * H_)    h2pp[idx] = (__bf16)0.0f;
  if (idx < 2)              bar[idx] = 0u;
}

// ---------------------------------------------------------------------------
// persistent LSTM layer.  Grid = NWG workgroups x 512 threads (16 waves).
// Wave (mi,gi): batch tile mi (16 rows), gate gi of this WG's 16 h-columns.
// LDS: W_ih slice, W_hh slice, double-buffered async-staged x_t, gate tiles, c.
// ---------------------------------------------------------------------------
__global__ void lstm_layer_kernel(const __bf16* __restrict__ xseq, int Din,
                                  const __bf16* __restrict__ WihP,
                                  const __bf16* __restrict__ WhhP,
                                  const float*  __restrict__ biasP,
                                  __bf16* __restrict__ hbase, int is_l0,
                                  unsigned* __restrict__ bcnt,
                                  unsigned* __restrict__ bgen) {
  const int s    = blockIdx.x;                 // slice 0..31
  const int wave = threadIdx.x >> 5;           // 0..15
  const int lane = threadIdx.x & 31;
  const int mi   = wave >> 2;                  // batch tile 0..3
  const int gi   = wave & 3;                   // gate i/f/g/o
  const int l16  = lane & 15;
  const int koff   = (lane < 16) ? 0 : 8;      // A-fragment K offset
  const int kboff  = (lane < 16) ? 0 : 16;     // B-fragment K offset
  const int rowoff = (lane < 16) ? 0 : 8;      // C-tile row offset
  const int arow = mi * 16 + l16;              // batch row for A fragments

  extern __shared__ char smem[];
  __bf16* sWih   = (__bf16*)smem;                       // [64][Din]
  __bf16* sWhh   = sWih + 64 * Din;                     // [64][H_]
  __bf16* sX     = sWhh + 64 * H_;                      // [2][64][Din]
  float*  sGates = (float*)(sX + 2 * 64 * Din);         // [4][64][16]
  float*  sC     = sGates + 4 * 64 * 16;                // [64][16]

  const int xcnt8 = 64 * Din / 8;              // b128 copies per x_t tile
  const unsigned sXaddr = (unsigned)(size_t)(void*)sX;  // LDS byte address
  const unsigned xbytes = (unsigned)(64 * Din * 2);

  // stage weight slices into LDS once; reused for all T steps
  {
    const v8bf* g0 = (const v8bf*)(WihP + (long long)s * 64 * Din);
    v8bf* l0p = (v8bf*)sWih;
    for (int i = threadIdx.x; i < 64 * Din / 8; i += blockDim.x) l0p[i] = g0[i];
    const v8bf* g1 = (const v8bf*)(WhhP + (long long)s * 64 * H_);
    v8bf* l1p = (v8bf*)sWhh;
    for (int i = threadIdx.x; i < 64 * H_ / 8; i += blockDim.x) l1p[i] = g1[i];
    for (int i = threadIdx.x; i < 64 * 16; i += blockDim.x) sC[i] = 0.0f;
  }
  // async-stage x_0 into buffer 0 (overlaps with the LDS weight staging above)
  for (int i = threadIdx.x; i < xcnt8; i += blockDim.x)
    async_ld_b128(sXaddr + (unsigned)i * 16u, xseq + (long long)i * 8);
  wait_async0();
  __syncthreads();

  const float bv = biasP[s * 64 + gi * 16 + l16];
  const __bf16* sWihCol = sWih + (gi * 16 + l16) * Din;  // this lane's B column
  const __bf16* sWhhCol = sWhh + (gi * 16 + l16) * H_;

  for (int t = 0; t < T_; ++t) {
    // kick off async DMA of x_{t+1} into the other LDS buffer; it proceeds
    // in the background while this step's WMMA work runs
    if (t + 1 < T_) {
      const __bf16* gx = xseq + (long long)(t + 1) * B_ * Din;
      const unsigned dst = sXaddr + (unsigned)((t + 1) & 1) * xbytes;
      for (int i = threadIdx.x; i < xcnt8; i += blockDim.x)
        async_ld_b128(dst + (unsigned)i * 16u, gx + (long long)i * 8);
    }

    const __bf16* xcur  = sX + (size_t)(t & 1) * 64 * Din;       // LDS
    const __bf16* hprev = hbase + (long long)(is_l0 ? t : (t & 1)) * B_ * H_;

    v8f acc = {bv, bv, bv, bv, bv, bv, bv, bv};

    // ---- x_t @ W_ih^T (A from LDS stage, B from LDS weights) ----
    const __bf16* arowx = xcur + (size_t)arow * Din;
#pragma unroll 4
    for (int k = 0; k < Din; k += 32) {
      v16bf a = load_a_frag(arowx, k, koff);
      v16bf b = load_b_frag(sWihCol, k, kboff);
      acc = __builtin_amdgcn_wmma_f32_16x16x32_bf16(false, a, false, b,
                                                    (short)0, acc, false, false);
    }
    // ---- h_{t-1} @ W_hh^T (A from global/L2 — true recurrence dep) ----
    const __bf16* arowh = hprev + (long long)arow * H_;
#pragma unroll 4
    for (int k = 0; k < H_; k += 32) {
      v16bf a = load_a_frag(arowh, k, koff);
      v16bf b = load_b_frag(sWhhCol, k, kboff);
      acc = __builtin_amdgcn_wmma_f32_16x16x32_bf16(false, a, false, b,
                                                    (short)0, acc, false, false);
    }

    // C-tile layout: VGPR r, lane l -> row mi*16 + r + rowoff, col l&15
#pragma unroll
    for (int r = 0; r < 8; ++r)
      sGates[(gi * 64 + mi * 16 + r + rowoff) * 16 + l16] = acc[r];
    __syncthreads();

    // elementwise cell update: 1024 (batch,col) pairs over 512 threads
    __bf16* hout = hbase + (long long)(is_l0 ? (t + 1) : ((t + 1) & 1)) * B_ * H_;
    for (int idx = threadIdx.x; idx < 64 * 16; idx += blockDim.x) {
      int b = idx >> 4, c = idx & 15;
      float ig = sGates[(0 * 64 + b) * 16 + c];
      float fg = sGates[(1 * 64 + b) * 16 + c];
      float gg = sGates[(2 * 64 + b) * 16 + c];
      float og = sGates[(3 * 64 + b) * 16 + c];
      float cn = sigmoidf_(fg) * sC[idx] + sigmoidf_(ig) * tanhf(gg);
      sC[idx] = cn;
      hout[(long long)b * H_ + s * 16 + c] = (__bf16)(sigmoidf_(og) * tanhf(cn));
    }
    wait_async0();                        // x_{t+1} stage complete
    grid_barrier(bcnt, bgen, (unsigned)gridDim.x);
  }
}

// ---------------------------------------------------------------------------
// final FC: out[64,256] = h2[64,512] @ fc_w^T + fc_b, one wave per 16x16 tile
// ---------------------------------------------------------------------------
__global__ void fc_kernel(const __bf16* __restrict__ h2,
                          const __bf16* __restrict__ fcw,   // [OUT][H] bf16
                          const float*  __restrict__ fcb,
                          float* __restrict__ out) {
  const int lane = threadIdx.x & 31;
  const int l16 = lane & 15;
  const int koff   = (lane < 16) ? 0 : 8;
  const int kboff  = (lane < 16) ? 0 : 16;
  const int rowoff = (lane < 16) ? 0 : 8;
  const int n0 = blockIdx.x * 16;
  const int m0 = blockIdx.y * 16;

  const float bv = fcb[n0 + l16];
  v8f acc = {bv, bv, bv, bv, bv, bv, bv, bv};
  const __bf16* arow = h2 + (long long)(m0 + l16) * H_;
  const __bf16* bcol = fcw + (long long)(n0 + l16) * H_;
#pragma unroll 4
  for (int k = 0; k < H_; k += 32) {
    v16bf a = load_a_frag(arow, k, koff);
    v16bf b = load_b_frag(bcol, k, kboff);
    acc = __builtin_amdgcn_wmma_f32_16x16x32_bf16(false, a, false, b,
                                                  (short)0, acc, false, false);
  }
#pragma unroll
  for (int r = 0; r < 8; ++r)
    out[(long long)(m0 + r + rowoff) * OUT_ + n0 + l16] = acc[r];
}

// ---------------------------------------------------------------------------
extern "C" void kernel_launch(void* const* d_in, const int* in_sizes, int n_in,
                              void* d_out, int out_size, void* d_ws, size_t ws_size,
                              hipStream_t stream) {
  const float* input = (const float*)d_in[0];
  const float* W_ih0 = (const float*)d_in[1];
  const float* W_hh0 = (const float*)d_in[2];
  const float* b_ih0 = (const float*)d_in[3];
  const float* b_hh0 = (const float*)d_in[4];
  const float* W_ih1 = (const float*)d_in[5];
  const float* W_hh1 = (const float*)d_in[6];
  const float* b_ih1 = (const float*)d_in[7];
  const float* b_hh1 = (const float*)d_in[8];
  const float* fc_w  = (const float*)d_in[9];
  const float* fc_b  = (const float*)d_in[10];
  float* out = (float*)d_out;

  char* ws = (char*)d_ws;
  size_t off = 0;
  auto take = [&](size_t bytes) -> char* {
    char* p = ws + off;
    off += (bytes + 255) & ~(size_t)255;
    return p;
  };
  unsigned* bar   = (unsigned*)take(2 * sizeof(unsigned));
  __bf16* x_bf    = (__bf16*)take((size_t)T_ * B_ * D_ * 2);
  __bf16* h1seq   = (__bf16*)take((size_t)(T_ + 1) * B_ * H_ * 2);
  __bf16* h2pp    = (__bf16*)take((size_t)2 * B_ * H_ * 2);
  __bf16* Wih0P   = (__bf16*)take((size_t)4 * H_ * D_ * 2);
  __bf16* Whh0P   = (__bf16*)take((size_t)4 * H_ * H_ * 2);
  __bf16* Wih1P   = (__bf16*)take((size_t)4 * H_ * H_ * 2);
  __bf16* Whh1P   = (__bf16*)take((size_t)4 * H_ * H_ * 2);
  float*  bias0P  = (float*)take((size_t)4 * H_ * 4);
  float*  bias1P  = (float*)take((size_t)4 * H_ * 4);
  __bf16* fcw_bf  = (__bf16*)take((size_t)OUT_ * H_ * 2);

  // LDS: weights + 2x x-stage + gates + c
  const size_t smem0 = (size_t)(64 * D_ + 64 * H_ + 2 * 64 * D_) * 2
                     + (4 * 64 * 16 + 64 * 16) * 4;                 // 184320 B
  const size_t smem1 = (size_t)(64 * H_ + 64 * H_ + 2 * 64 * H_) * 2
                     + (4 * 64 * 16 + 64 * 16) * 4;                 // 282624 B
  (void)hipFuncSetAttribute((const void*)lstm_layer_kernel,
                            hipFuncAttributeMaxDynamicSharedMemorySize,
                            (int)(smem1 > smem0 ? smem1 : smem0));

  // ---- prep ----
  {
    long long n = (long long)B_ * T_ * D_;
    convert_x_kernel<<<(unsigned)((n + 255) / 256), 256, 0, stream>>>(input, x_bf);
  }
  pack_gate_slices_kernel<<<(4 * H_ * D_) / 256, 256, 0, stream>>>(W_ih0, Wih0P, D_);
  pack_gate_slices_kernel<<<(4 * H_ * H_) / 256, 256, 0, stream>>>(W_hh0, Whh0P, H_);
  pack_gate_slices_kernel<<<(4 * H_ * H_) / 256, 256, 0, stream>>>(W_ih1, Wih1P, H_);
  pack_gate_slices_kernel<<<(4 * H_ * H_) / 256, 256, 0, stream>>>(W_hh1, Whh1P, H_);
  pack_bias_kernel<<<(4 * H_) / 256, 256, 0, stream>>>(b_ih0, b_hh0, bias0P);
  pack_bias_kernel<<<(4 * H_) / 256, 256, 0, stream>>>(b_ih1, b_hh1, bias1P);
  convert_fcw_kernel<<<(OUT_ * H_) / 256, 256, 0, stream>>>(fc_w, fcw_bf);
  zero_init_kernel<<<(2 * B_ * H_ + 255) / 256, 256, 0, stream>>>(h1seq, h2pp, bar);

  // ---- layer 0: persistent recurrence, h1seq[t+1] = h1(t) ----
  lstm_layer_kernel<<<NWG, 512, smem0, stream>>>(x_bf, D_, Wih0P, Whh0P, bias0P,
                                                 h1seq, 1, bar, bar + 1);
  // ---- layer 1: consumes h1seq (shifted by the zero block), ping-pong h2 ----
  lstm_layer_kernel<<<NWG, 512, smem1, stream>>>(h1seq + B_ * H_, H_, Wih1P, Whh1P,
                                                 bias1P, h2pp, 0, bar, bar + 1);
  // final h2 lands in slot (T_ & 1) == 0
  fc_kernel<<<dim3(OUT_ / 16, B_ / 16), 32, 0, stream>>>(h2pp, fcw_bf, fc_b, out);
}
